// InteractiveWindowedSelfAttention_10986526343363
// MI455X (gfx1250) — matrix operand order repaired and checked by reading, not compiled, not run on hardware
//
#include <hip/hip_runtime.h>
#include <hip/hip_bf16.h>

// ---------------------------------------------------------------------------
// InteractiveWindowedSelfAttention for MI455X (gfx1250, wave32, WMMA)
// Precision: bf16 activations/weights, fp32 WMMA accumulate, fp32 output.
// All GEMM-shaped work goes through v_wmma_f32_16x16x32_bf16.
// LDS tiles are stored pre-swizzled in fragment order so every WMMA operand
// is built from two contiguous ds_load_b128's (no scalar gathers).
// 3x3 conv weights are pre-transposed to tap-major so staging is vectorized.
// ---------------------------------------------------------------------------

typedef __attribute__((ext_vector_type(16))) __bf16 bf16x16;
typedef __attribute__((ext_vector_type(8)))  __bf16 bf16x8;
typedef __attribute__((ext_vector_type(2)))  __bf16 bf16x2;
typedef __attribute__((ext_vector_type(8)))  float  f32x8;

#define BATCH 16
#define CHN   256
#define IMH   128
#define IMW   128
#define HWSZ  (IMH * IMW)
#define XSTR  40          // padded LDS row stride (bf16) -> 20 dwords, conflict-free

static __device__ __forceinline__ bf16x16 join16(bf16x8 lo, bf16x8 hi) {
    return __builtin_shufflevector(lo, hi, 0,1,2,3,4,5,6,7,8,9,10,11,12,13,14,15);
}
static __device__ __forceinline__ f32x8 wmma_bf16(bf16x16 a, bf16x16 b, f32x8 c) {
    // (neg_a, A, neg_b, B, c_mod, C, reuse_a, reuse_b)
    return __builtin_amdgcn_wmma_f32_16x16x32_bf16(false, a, false, b, (short)0, c, false, false);
}

// ---------------------------------------------------------------------------
// fp32 -> bf16 convert (weights)
// ---------------------------------------------------------------------------
__global__ void __launch_bounds__(256) cvt_kernel(const float* __restrict__ in,
                                                  __bf16* __restrict__ out, int n) {
    int i = blockIdx.x * 256 + threadIdx.x;
    if (i < n) out[i] = (__bf16)in[i];
}

// local_w [o][c][3][3] fp32 -> LWT [tap][o][c] bf16 (tap-major, c-contiguous)
__global__ void __launch_bounds__(256) cvt_lw_kernel(const float* __restrict__ in,
                                                     __bf16* __restrict__ out) {
    int i = blockIdx.x * 256 + threadIdx.x;      // over 9*256*256
    int tap = i / (CHN * CHN);
    int rem = i % (CHN * CHN);
    int o = rem / CHN, c = rem % CHN;
    out[i] = (__bf16)in[((size_t)o * CHN + c) * 9 + tap];
}

// ---------------------------------------------------------------------------
// Channel LayerNorm: x [B][C][HW] fp32 -> xn bf16 (same layout).
// ---------------------------------------------------------------------------
__global__ void __launch_bounds__(256) ln_kernel(const float* __restrict__ x,
                                                 const float* __restrict__ g,
                                                 const float* __restrict__ bb,
                                                 __bf16* __restrict__ xn) {
    int p  = blockIdx.x * 256 + threadIdx.x;   // 0 .. B*HW-1
    int b  = p >> 14;
    int hw = p & (HWSZ - 1);
    const float* xp = x + (size_t)b * CHN * HWSZ + hw;
    float s = 0.f, s2 = 0.f;
    for (int c = 0; c < CHN; ++c) {
        float v = xp[(size_t)c * HWSZ];
        s += v; s2 += v * v;
    }
    float mean = s * (1.f / CHN);
    float var  = s2 * (1.f / CHN) - mean * mean;
    float inv  = rsqrtf(var + 1e-5f);
    __bf16* op = xn + (size_t)b * CHN * HWSZ + hw;
    for (int c = 0; c < CHN; ++c) {
        float v = xp[(size_t)c * HWSZ];
        op[(size_t)c * HWSZ] = (__bf16)(((v - mean) * inv) * g[c] + bb[c]);
    }
}

// ---------------------------------------------------------------------------
// GEMM: Y[b][o][p] = sum_c W[o][c] * X[b][c][p]   (Cin = Cout = 256)
// Block: 256 thr = 8 waves; tile M=64 (out ch) x N=128 (pixels), K-step 32.
// Xs stored transposed [pixel][k]; staging packs channel pairs -> b32 stores.
// ---------------------------------------------------------------------------
template <bool FINAL>
__global__ void __launch_bounds__(256) gemm_wmma(const __bf16* __restrict__ Wmat,
                                                 const __bf16* __restrict__ X,
                                                 __bf16* __restrict__ Ybf,
                                                 float* __restrict__ Yf,
                                                 const float* __restrict__ bias) {
    __shared__ __bf16 Xs[128 * XSTR];  // [p][k], padded
    __shared__ __bf16 Ws[64 * 32];     // [o][k]
    const int t    = threadIdx.x;
    const int p0   = blockIdx.x * 128;
    const int o0   = blockIdx.y * 64;
    const int b    = blockIdx.z;
    const int lane = t & 31, wv = t >> 5;
    const int hf   = lane >> 4, l15 = lane & 15;
    const int mt   = wv & 3;
    const int ng   = (wv >> 2) * 4;    // n-tile base: 0 or 4

    f32x8 acc[4];
#pragma unroll
    for (int j = 0; j < 4; ++j)
#pragma unroll
        for (int e = 0; e < 8; ++e) acc[j][e] = 0.f;

    const size_t xbase = (size_t)b * CHN * HWSZ + p0;
    const int kk  = (t & 15) * 2;      // channel pair
    const int seg = (t >> 4) * 8;      // 8-pixel segment
    for (int k0 = 0; k0 < CHN; k0 += 32) {
        {   // stage X tile transposed: 2 channel rows x 8 pixels per thread
            const __bf16* src = X + xbase + (size_t)(k0 + kk) * HWSZ + seg;
            bf16x8 a0 = *(const bf16x8*)(src);
            bf16x8 a1 = *(const bf16x8*)(src + HWSZ);
#pragma unroll
            for (int e = 0; e < 8; ++e) {
                bf16x2 pr; pr[0] = a0[e]; pr[1] = a1[e];
                *(bf16x2*)(Xs + (seg + e) * XSTR + kk) = pr;
            }
            // speculative prefetch of next K-chunk (dropped if invalid)
            __builtin_prefetch(src + (size_t)32 * HWSZ, 0, 1);
        }
        {   // stage W tile: 2048 bf16, 8 contiguous per thread
            int base = t * 8;
            int orow = base >> 5, col = base & 31;
            *(bf16x8*)(Ws + base) =
                *(const bf16x8*)(Wmat + (size_t)(o0 + orow) * CHN + k0 + col);
        }
        __syncthreads();

        // A fragment: row = mt*16 + l15, K elems {8*hf.., 16+8*hf..}
        bf16x16 afrag;
        {
            const __bf16* r = Ws + (mt * 16 + l15) * 32 + hf * 8;
            afrag = join16(*(const bf16x8*)r, *(const bf16x8*)(r + 16));
        }
#pragma unroll
        for (int j = 0; j < 4; ++j) {
            int nt = ng + j;
            // B fragment: col = nt*16 + l15, K elems {8*hf.., 16+8*hf..}
            const __bf16* r = Xs + (nt * 16 + l15) * XSTR + hf * 8;
            bf16x16 bfrag = join16(*(const bf16x8*)r, *(const bf16x8*)(r + 16));
            acc[j] = wmma_bf16(afrag, bfrag, acc[j]);
        }
        __syncthreads();
    }

#pragma unroll
    for (int j = 0; j < 4; ++j) {
        int n = (ng + j) * 16 + l15;
#pragma unroll
        for (int v = 0; v < 8; ++v) {
            int m = o0 + mt * 16 + v + hf * 8;
            size_t idx = ((size_t)b * CHN + m) * HWSZ + p0 + n;
            float val = acc[j][v];
            if (FINAL) Yf[idx]  = val + bias[m];
            else       Ybf[idx] = (__bf16)val;
        }
    }
}

// ---------------------------------------------------------------------------
// 3x3 conv on V: per source row (dy) and 32-channel chunk, stage the row
// strip ONCE with +-1 pixel halo AND all three dx weight slices, then run
// the 3 dx taps against the fully resident tiles (B pointer shifts by one
// LDS row per tap).  Halo columns are zeroed once (staging never touches
// rows 0/129).  Weights come from tap-major LWT -> vector staging.
// Fused epilogue: SUM[o][p] = bf16(conv + local_b[o] + SUM[o][p] (attn out)).
// ---------------------------------------------------------------------------
__global__ void __launch_bounds__(256) conv3_wmma(const __bf16* __restrict__ Vb,
                                                  const __bf16* __restrict__ LWT, // [tap][o][c]
                                                  const float* __restrict__ lbias,
                                                  __bf16* __restrict__ SUM) {
    __shared__ __bf16 Xs[130 * XSTR];   // [x+1][k] with halo rows 0 and 129
    __shared__ __bf16 Ws3[3 * 64 * 32]; // [dx][o][k]
    const int t    = threadIdx.x;
    const int y    = blockIdx.x;
    const int o0   = blockIdx.y * 64;
    const int b    = blockIdx.z;
    const int lane = t & 31, wv = t >> 5;
    const int hf   = lane >> 4, l15 = lane & 15;
    const int mt   = wv & 3;
    const int ng   = (wv >> 2) * 4;

    // zero halo columns once; visible after the first barrier below
    if (t < XSTR) {
        Xs[t]               = (__bf16)0.f;
        Xs[129 * XSTR + t]  = (__bf16)0.f;
    }

    f32x8 acc[4];
#pragma unroll
    for (int j = 0; j < 4; ++j)
#pragma unroll
        for (int e = 0; e < 8; ++e) acc[j][e] = 0.f;

    const int kk  = (t & 15) * 2;      // channel pair
    const int seg = (t >> 4) * 8;      // 8-pixel segment
    for (int dyi = 0; dyi < 3; ++dyi) {
        int sy = y + dyi - 1;
        bool rowok = (unsigned)sy < (unsigned)IMH;
        for (int k0 = 0; k0 < CHN; k0 += 32) {
            if (rowok) {   // stage shifted-row strip, transposed
                const __bf16* src =
                    Vb + ((size_t)b * CHN + k0 + kk) * HWSZ + sy * IMW + seg;
                bf16x8 a0 = *(const bf16x8*)(src);
                bf16x8 a1 = *(const bf16x8*)(src + HWSZ);
#pragma unroll
                for (int e = 0; e < 8; ++e) {
                    bf16x2 pr; pr[0] = a0[e]; pr[1] = a1[e];
                    *(bf16x2*)(Xs + (1 + seg + e) * XSTR + kk) = pr;
                }
                __builtin_prefetch(src + (size_t)32 * HWSZ, 0, 1);
            } else {
                bf16x2 z; z[0] = (__bf16)0.f; z[1] = (__bf16)0.f;
#pragma unroll
                for (int e = 0; e < 8; ++e)
                    *(bf16x2*)(Xs + (1 + seg + e) * XSTR + kk) = z;
            }
            {   // stage 3 dx weight slices, contiguous vector loads
                int base = t * 8;
                int orow = base >> 5, col = base & 31;
#pragma unroll
                for (int dxi = 0; dxi < 3; ++dxi) {
                    *(bf16x8*)(Ws3 + dxi * 2048 + base) = *(const bf16x8*)(
                        LWT + ((size_t)(dyi * 3 + dxi) * CHN + o0 + orow) * CHN +
                        k0 + col);
                }
            }
            __syncthreads();

#pragma unroll
            for (int dxi = 0; dxi < 3; ++dxi) {
                bf16x16 afrag;
                {
                    const __bf16* r =
                        Ws3 + dxi * 2048 + (mt * 16 + l15) * 32 + hf * 8;
                    afrag = join16(*(const bf16x8*)r, *(const bf16x8*)(r + 16));
                }
#pragma unroll
                for (int j = 0; j < 4; ++j) {
                    int nt = ng + j;
                    const __bf16* r =
                        Xs + (nt * 16 + l15 + dxi) * XSTR + hf * 8;
                    bf16x16 bfrag =
                        join16(*(const bf16x8*)r, *(const bf16x8*)(r + 16));
                    acc[j] = wmma_bf16(afrag, bfrag, acc[j]);
                }
            }
            __syncthreads();
        }
    }

    const int p0 = y * IMW;
#pragma unroll
    for (int j = 0; j < 4; ++j) {
        int n = (ng + j) * 16 + l15;
#pragma unroll
        for (int v = 0; v < 8; ++v) {
            int m = o0 + mt * 16 + v + hf * 8;
            size_t idx = ((size_t)b * CHN + m) * HWSZ + p0 + n;
            float val = acc[j][v] + lbias[m] + (float)SUM[idx];
            SUM[idx] = (__bf16)val;
        }
    }
}

// ---------------------------------------------------------------------------
// Windowed attention: one block per 8x8 window, one wave per head.
// Qs/Ks: [head][t][d] (fragment-contiguous for QK^T); Vs: TRANSPOSED
// [head][d][t] so the P.V B-operand is also two contiguous b128 loads and
// its staging store is a single b128 per 8 pixels.
// P spilled to LDS (aliases dead Q/K space after a barrier).
// LDS: dynamic 96 KB.
// ---------------------------------------------------------------------------
__global__ void __launch_bounds__(256) winattn_wmma(const __bf16* __restrict__ Q,
                                                    const __bf16* __restrict__ Kb,
                                                    const __bf16* __restrict__ Vb,
                                                    __bf16* __restrict__ OUT) {
    extern __shared__ __bf16 smem[];            // 3 * 8 * 64 * 32 bf16 = 96 KB
    __bf16* Qs  = smem;                         // [head][t][d]
    __bf16* Ks  = smem + 8 * 64 * 32;           // [head][t][d]
    __bf16* VsT = smem + 2 * 8 * 64 * 32;       // [head][d][t]

    const int t    = threadIdx.x;
    const int lane = t & 31, wv = t >> 5;       // wv == head
    const int hf   = lane >> 4, l15 = lane & 15;
    const int win  = blockIdx.x;
    const int b    = win >> 8;
    const int wy   = (win >> 4) & 15, wx = win & 15;
    const int basepix = (wy * 8) * IMW + wx * 8;

    // Phase 0: stage Q/K/V for all 8 heads
    for (int i = t; i < 2048; i += 256) {
        int ch = i >> 3, irow = i & 7;
        size_t g = ((size_t)b * CHN + ch) * HWSZ + basepix + irow * IMW;
        int head = ch >> 5, d = ch & 31;
        bf16x8 q8 = *(const bf16x8*)(Q  + g);
        bf16x8 k8 = *(const bf16x8*)(Kb + g);
        bf16x8 v8 = *(const bf16x8*)(Vb + g);
        int qb = head * 2048 + d;
#pragma unroll
        for (int j = 0; j < 8; ++j) {
            int r = (irow * 8 + j) * 32;
            Qs[qb + r] = q8[j];
            Ks[qb + r] = k8[j];
        }
        // V transposed: contiguous along t
        *(bf16x8*)(VsT + head * 2048 + d * 64 + irow * 8) = v8;
    }
    __syncthreads();

    const __bf16* Qh  = Qs  + wv * 2048;
    const __bf16* Kh  = Ks  + wv * 2048;
    const __bf16* VhT = VsT + wv * 2048;

    // Phase 1: dots[64][64] = Q K^T  (head dim 32 == one WMMA K)
    f32x8 dots[4][4];
#pragma unroll
    for (int mi = 0; mi < 4; ++mi)
#pragma unroll
        for (int ni = 0; ni < 4; ++ni)
#pragma unroll
            for (int e = 0; e < 8; ++e) dots[mi][ni][e] = 0.f;

#pragma unroll
    for (int mi = 0; mi < 4; ++mi) {
        bf16x16 a;
        {
            const __bf16* r = Qh + (mi * 16 + l15) * 32 + hf * 8;
            a = join16(*(const bf16x8*)r, *(const bf16x8*)(r + 16));
        }
#pragma unroll
        for (int ni = 0; ni < 4; ++ni) {
            const __bf16* r = Kh + (ni * 16 + l15) * 32 + hf * 8;
            bf16x16 bf_ = join16(*(const bf16x8*)r, *(const bf16x8*)(r + 16));
            dots[mi][ni] = wmma_bf16(a, bf_, dots[mi][ni]);
        }
    }

    // Phase 2: softmax over key dim (row = mi*16 + v + 8*hf, cols across l15)
    const float scale = 0.17677669529663687f;   // 1/sqrt(32)
    float rowinv[4][8];
#pragma unroll
    for (int mi = 0; mi < 4; ++mi) {
#pragma unroll
        for (int v = 0; v < 8; ++v) {
            float mx = dots[mi][0][v];
            mx = fmaxf(mx, dots[mi][1][v]);
            mx = fmaxf(mx, dots[mi][2][v]);
            mx = fmaxf(mx, dots[mi][3][v]);
            mx *= scale;
#pragma unroll
            for (int s = 1; s < 16; s <<= 1) mx = fmaxf(mx, __shfl_xor(mx, s, 32));
            float sum = 0.f;
#pragma unroll
            for (int ni = 0; ni < 4; ++ni) {
                float p = __expf(dots[mi][ni][v] * scale - mx);
                dots[mi][ni][v] = p;
                sum += p;
            }
#pragma unroll
            for (int s = 1; s < 16; s <<= 1) sum += __shfl_xor(sum, s, 32);
            rowinv[mi][v] = 1.f / sum;
        }
    }

    __syncthreads();   // all waves done reading Qs/Ks -> safe to alias as P

    // Phase 3: spill P (bf16) to LDS: head h gets smem + h*4096, [64][64]
    __bf16* Ph = smem + wv * 4096;
#pragma unroll
    for (int mi = 0; mi < 4; ++mi)
#pragma unroll
        for (int ni = 0; ni < 4; ++ni)
#pragma unroll
            for (int v = 0; v < 8; ++v)
                Ph[(mi * 16 + v + 8 * hf) * 64 + ni * 16 + l15] =
                    (__bf16)dots[mi][ni][v];
    __syncthreads();

    // Phase 4: out[64][32] = P V  (K = 64 keys, two K-steps)
    f32x8 outacc[4][2];
#pragma unroll
    for (int mi = 0; mi < 4; ++mi)
#pragma unroll
        for (int ni = 0; ni < 2; ++ni)
#pragma unroll
            for (int e = 0; e < 8; ++e) outacc[mi][ni][e] = 0.f;

#pragma unroll
    for (int k0 = 0; k0 < 64; k0 += 32) {
#pragma unroll
        for (int mi = 0; mi < 4; ++mi) {
            bf16x16 a;
            {
                const __bf16* r = Ph + (mi * 16 + l15) * 64 + k0 + hf * 8;
                a = join16(*(const bf16x8*)r, *(const bf16x8*)(r + 16));
            }
#pragma unroll
            for (int ni = 0; ni < 2; ++ni) {
                // B[k=key j][n=d] = V[j][d] = VhT[d][j]; contiguous along j
                const __bf16* r = VhT + (ni * 16 + l15) * 64 + k0 + hf * 8;
                bf16x16 bf_ = join16(*(const bf16x8*)r, *(const bf16x8*)(r + 16));
                outacc[mi][ni] = wmma_bf16(a, bf_, outacc[mi][ni]);
            }
        }
    }

    // Phase 5: normalize rows and scatter to [b][head*32+d][pixel]
#pragma unroll
    for (int mi = 0; mi < 4; ++mi) {
#pragma unroll
        for (int ni = 0; ni < 2; ++ni) {
#pragma unroll
            for (int v = 0; v < 8; ++v) {
                int trow = mi * 16 + v + 8 * hf;
                int d    = ni * 16 + l15;
                int ch   = wv * 32 + d;
                int pix  = basepix + (trow >> 3) * IMW + (trow & 7);
                OUT[((size_t)b * CHN + ch) * HWSZ + pix] =
                    (__bf16)(outacc[mi][ni][v] * rowinv[mi][v]);
            }
        }
    }
}

// ---------------------------------------------------------------------------
// Host launcher
// ---------------------------------------------------------------------------
extern "C" void kernel_launch(void* const* d_in, const int* in_sizes, int n_in,
                              void* d_out, int out_size, void* d_ws, size_t ws_size,
                              hipStream_t stream) {
    const float* x  = (const float*)d_in[0];
    const float* g  = (const float*)d_in[1];
    const float* bb = (const float*)d_in[2];
    const float* wq = (const float*)d_in[3];
    const float* wk = (const float*)d_in[4];
    const float* wv = (const float*)d_in[5];
    const float* lw = (const float*)d_in[6];
    const float* lb = (const float*)d_in[7];
    const float* ow = (const float*)d_in[8];
    const float* ob = (const float*)d_in[9];

    char* ws = (char*)d_ws;
    const size_t tsz = (size_t)BATCH * CHN * HWSZ * sizeof(__bf16);  // 134 MB
    __bf16* XN  = (__bf16*)(ws + 0 * tsz);
    __bf16* Qb  = (__bf16*)(ws + 1 * tsz);
    __bf16* Kb  = (__bf16*)(ws + 2 * tsz);
    __bf16* Vbuf= (__bf16*)(ws + 3 * tsz);
    __bf16* SM  = (__bf16*)(ws + 4 * tsz);
    __bf16* WQB = (__bf16*)(ws + 5 * tsz);
    __bf16* WKB = WQB + CHN * CHN;
    __bf16* WVB = WKB + CHN * CHN;
    __bf16* LWT = WVB + CHN * CHN;       // [tap][o][c]
    __bf16* OWB = LWT + CHN * CHN * 9;

    // weights -> bf16 (local_w also transposed to tap-major)
    cvt_kernel<<<(CHN * CHN + 255) / 256, 256, 0, stream>>>(wq, WQB, CHN * CHN);
    cvt_kernel<<<(CHN * CHN + 255) / 256, 256, 0, stream>>>(wk, WKB, CHN * CHN);
    cvt_kernel<<<(CHN * CHN + 255) / 256, 256, 0, stream>>>(wv, WVB, CHN * CHN);
    cvt_lw_kernel<<<(CHN * CHN * 9 + 255) / 256, 256, 0, stream>>>(lw, LWT);
    cvt_kernel<<<(CHN * CHN + 255) / 256, 256, 0, stream>>>(ow, OWB, CHN * CHN);

    // channel layernorm -> bf16
    ln_kernel<<<BATCH * HWSZ / 256, 256, 0, stream>>>(x, g, bb, XN);

    // q/k/v projections
    dim3 gg(HWSZ / 128, CHN / 64, BATCH);
    gemm_wmma<false><<<gg, 256, 0, stream>>>(WQB, XN, Qb, nullptr, nullptr);
    gemm_wmma<false><<<gg, 256, 0, stream>>>(WKB, XN, Kb, nullptr, nullptr);
    gemm_wmma<false><<<gg, 256, 0, stream>>>(WVB, XN, Vbuf, nullptr, nullptr);

    // windowed attention -> SM
    winattn_wmma<<<BATCH * 16 * 16, 256, 3 * 8 * 64 * 32 * sizeof(__bf16), stream>>>(
        Qb, Kb, Vbuf, SM);

    // 3x3 conv on V, fused add of attention output into SM
    dim3 gc(IMH, CHN / 64, BATCH);
    conv3_wmma<<<gc, 256, 0, stream>>>(Vbuf, LWT, lb, SM);

    // final 1x1 conv + bias -> fp32 d_out
    gemm_wmma<true><<<gg, 256, 0, stream>>>(OWB, SM, nullptr, (float*)d_out, ob);
}
